// SlicedVectorQuantizeEMA_3272765079615
// MI455X (gfx1250) — matrix operand order repaired and checked by reading, not compiled, not run on hardware
//
#include <hip/hip_runtime.h>
#include <hip/hip_bf16.h>

// ---------------------------------------------------------------------------
// SlicedVectorQuantizeEMA for MI455X (gfx1250, wave32, WMMA + TDM)
//
// x (32,512,2048) f32; per-half: fin (65536,256), emb (512,256), K=512 codes.
// Distance GEMM runs on v_wmma_f32_16x16x32_bf16 with the bf16 codebook
// staged into LDS by the Tensor Data Mover (tensor_load_to_lds, TENSORcnt),
// using TDM pad_enable to insert 16B/row padding (row stride 132 dwords) so
// the per-lane ds_load_b128 B-fragment reads spread across LDS banks.
// All EMA statistics stay fp32. Problem is memory bound (~0.55 GB @ 23.3TB/s).
// ---------------------------------------------------------------------------

typedef __attribute__((ext_vector_type(16))) __bf16 v16bf;
typedef __attribute__((ext_vector_type(8)))  float  v8f;
typedef __attribute__((ext_vector_type(4)))  unsigned int v4u;
typedef __attribute__((ext_vector_type(8)))  int    v8i;
typedef __attribute__((ext_vector_type(4)))  int    v4i;

#define VQ_K      512
#define VQ_SUBD   256
#define VQ_NROWS  65536      // 32 * 2048
#define VQ_B      32
#define VQ_T      2048
#define VQ_DECAY  0.99f
#define VQ_BETA   0.25f
#define VQ_EPS    1e-05f

// LDS staging: 256 codes/chunk, row stride 256 + 8 bf16 pad = 264 elems
#define VQ_CHUNK_CODES 256
#define VQ_SROW        264                    // bf16 elems per padded row
#define VQ_SB_ELEMS    (VQ_CHUNK_CODES * VQ_SROW)   // 67584 elems = 132 KB

// ---------------- workspace layout (byte offsets into d_ws) ----------------
#define WS_EMBB1   ((size_t)0)                        // 512*256 bf16 = 262144 B
#define WS_EMBB2   (WS_EMBB1 + 262144)
#define WS_ENORM1  (WS_EMBB2 + 262144)                // 512 f32
#define WS_ENORM2  (WS_ENORM1 + 2048)
#define WS_IDX1    (WS_ENORM2 + 2048)                 // 65536 i32
#define WS_IDX2    (WS_IDX1 + 262144)
#define WS_COUNTS1 (WS_IDX2 + 262144)                 // 512 f32 (zeroed)
#define WS_COUNTS2 (WS_COUNTS1 + 2048)
#define WS_DW1     (WS_COUNTS2 + 2048)                // 512*256 f32 (zeroed)
#define WS_DW2     (WS_DW1 + 524288)
#define WS_NEWEMB1 (WS_DW2 + 524288)                  // 512*256 f32
#define WS_NEWEMB2 (WS_NEWEMB1 + 524288)
#define WS_SCALARS (WS_NEWEMB2 + 524288)              // [loss, perp1, perp2] (zeroed)
#define WS_ZERO_BEG WS_COUNTS1
#define WS_ZERO_LEN (WS_DW2 + 524288 - WS_COUNTS1)

// ---------------------------------------------------------------------------
// Kernel 1: codebook -> bf16 copy + per-code squared norms (tiny: 512 KB)
// ---------------------------------------------------------------------------
__global__ __launch_bounds__(128) void vq_prep_kernel(
    const float* __restrict__ emb1, const float* __restrict__ emb2,
    __bf16* __restrict__ embB1, __bf16* __restrict__ embB2,
    float* __restrict__ enorm1, float* __restrict__ enorm2)
{
    const int h = blockIdx.y;
    const float* e  = h ? emb2  : emb1;
    __bf16*      eb = h ? embB2 : embB1;
    float*       en = h ? enorm2 : enorm1;

    const int k = blockIdx.x * 128 + threadIdx.x;   // code id
    if (k >= VQ_K) return;
    float s = 0.0f;
    for (int c = 0; c < VQ_SUBD; ++c) {
        float v = e[(size_t)k * VQ_SUBD + c];
        s += v * v;
        eb[(size_t)k * VQ_SUBD + c] = (__bf16)v;
    }
    en[k] = s;
}

// ---------------------------------------------------------------------------
// TDM staging: DMA 256 codebook rows (128 u32 each) into LDS with 4-dword
// padding after every 128 dwords (pad_interval code 6 = 128, pad_amount
// code 3 = 4 dwords) -> LDS row stride 132 dwords = VQ_SROW bf16.
// D# bitfields per CDNA5 ISA ch.8 (group0: count/lds/global/type,
// group1: data_size/pad/tensor dims/tile dims/stride).
// This toolchain exposes the 6-arg builtin:
//   (uint32x4 g0, int32x8 g1, int32x4 g2, int32x4 g3, int32x8 extra, i32 cpol)
// ---------------------------------------------------------------------------
__device__ __forceinline__ void vq_stage_chunk(const __bf16* gsrc,
                                               __bf16* sB, int tid)
{
#if __has_builtin(__builtin_amdgcn_tensor_load_to_lds)
    if (tid < 32) {   // wave 0 issues the DMA (TDM ignores EXEC; per-wave op)
        const unsigned long long ga = (unsigned long long)(uintptr_t)gsrc;
        const unsigned lds_addr = (unsigned)(uintptr_t)sB;   // low 32b = LDS offset

        v4u g0;
        g0[0] = 1u;                                   // count=1 valid descriptor
        g0[1] = lds_addr;                             // lds_addr (bytes)
        g0[2] = (unsigned)ga;                         // global_addr[31:0]
        g0[3] = (unsigned)((ga >> 32) & 0x01FFFFFFu)  // global_addr[56:32]
                | 0x80000000u;                        // type=2 ("image")

        v8i g1;
        g1[0] = (int)((2u << 16)        // data_size = 4 bytes
                    | (1u << 20)        // pad_enable
                    | (6u << 22)        // pad_interval: code 6 = 128 dwords
                    | (3u << 25));      // pad_amount:  code 3 = 4 dwords
        g1[1] = (int)(128u << 16);      // tensor_dim0 = 128 u32 (bits 79:48 lo)
        g1[2] = (int)(256u << 16);      // tensor_dim0 hi=0 | tensor_dim1 = 256
        g1[3] = (int)(128u << 16);      // tensor_dim1 hi=0 | tile_dim0 = 128
        g1[4] = 256;                    // tile_dim1 = 256, tile_dim2 = 0
        g1[5] = 128;                    // tensor_dim0_stride = 128 u32
        g1[6] = 0;                      // stride hi / tensor_dim1_stride lo
        g1[7] = 0;

        v4i z4 = {0, 0, 0, 0};          // groups 2/3 unused (2-D tile)
        v8i z8 = {0, 0, 0, 0, 0, 0, 0, 0};
        __builtin_amdgcn_tensor_load_to_lds(g0, g1, z4, z4, z8, 0);
#if __has_builtin(__builtin_amdgcn_s_wait_tensorcnt)
        __builtin_amdgcn_s_wait_tensorcnt(0);
#else
        asm volatile("s_wait_tensorcnt 0x0" ::: "memory");
#endif
    }
#else
    // Fallback: cooperative padded copy (same LDS layout as the TDM path).
    const unsigned* src = (const unsigned*)gsrc;
    unsigned* dst = (unsigned*)sB;
    for (int i = tid; i < VQ_CHUNK_CODES * 128; i += 512) {
        const int k  = i >> 7;          // code row
        const int dd = i & 127;         // dword within row
        __builtin_prefetch(src + i + 2048, 0, 0);
        dst[k * 132 + dd] = src[i];
    }
#endif
}

// ---------------------------------------------------------------------------
// Kernel 2: WMMA argmin. One wave = 16 rows x 512 codes; 16 waves/block.
//   score[m][n] = sum_k fin[m,k] * emb[n,k]  (A = fin tile, B = emb^T tile)
//   dist = ||e_n||^2 - 2*score  (||f||^2 dropped: constant per row)
// A fragment layout (16-bit A 16x32, K order chosen here, not the ISA table order):
//   lane L: m = L&15, g = L>>4; vgpr j holds K = 16g + 2j, +1 (contiguous, matches B)
// B fragment layout (16-bit B 32x16):
//   lane L: n = L&15, g = L>>4; element e holds K = 16*g + e  (contiguous)
// C/D layout: vgpr r -> row M = r + 8g, column N = L&15.
// B is read from the TDM-staged LDS image (row stride 132 dwords).
// ---------------------------------------------------------------------------
__global__ __launch_bounds__(512) void vq_argmin_kernel(
    const float* __restrict__ x,
    const __bf16* __restrict__ embB1, const __bf16* __restrict__ embB2,
    const float* __restrict__ enorm1, const float* __restrict__ enorm2,
    int* __restrict__ idx1, int* __restrict__ idx2,
    float* __restrict__ counts1, float* __restrict__ counts2)
{
    __shared__ __bf16 sB[VQ_SB_ELEMS];          // 132 KB staging buffer

    const int h = blockIdx.y;
    const __bf16* embB  = h ? embB2  : embB1;
    const float*  enorm = h ? enorm2 : enorm1;
    int*          idxo  = h ? idx2   : idx1;
    float*        cnts  = h ? counts2 : counts1;

    const int tid   = threadIdx.x;
    const int lane  = tid & 31;
    const int wv    = tid >> 5;                 // 16 waves, 16 rows each
    const int rowbase = (blockIdx.x * 16 + wv) * 16;

    const int m = lane & 15;      // A row within tile / B column
    const int g = lane >> 4;      // lane half-group

    // fin[row, k] = x[b, h*256 + k, t], row = b*2048 + t
    const int row = rowbase + m;
    const int b   = row >> 11;
    const int t   = row & 2047;
    const float* xcol = x + ((size_t)(b * 512 + h * VQ_SUBD) * VQ_T + t);

    // Build all A fragments (16 rows x 256 K, f32 -> bf16) once.
    v16bf afrag[8];
    #pragma unroll
    for (int kc = 0; kc < 8; ++kc) {
        #pragma unroll
        for (int j = 0; j < 8; ++j) {
            const int kb = 16 * g + 2 * j;
            const int k0 = kc * 32 + kb;
            afrag[kc][2 * j]     = (__bf16)xcol[(size_t)k0 * VQ_T];
            afrag[kc][2 * j + 1] = (__bf16)xcol[(size_t)(k0 + 1) * VQ_T];
        }
    }

    float best[8];
    int   bidx[8];
    #pragma unroll
    for (int r = 0; r < 8; ++r) { best[r] = 3.4e38f; bidx[r] = 0; }

    for (int chunk = 0; chunk < 2; ++chunk) {
        // Stage 256 codes into LDS (TDM DMA or cooperative fallback).
        vq_stage_chunk(embB + (size_t)chunk * VQ_CHUNK_CODES * VQ_SUBD, sB, tid);
        __syncthreads();

        for (int ntl = 0; ntl < 16; ++ntl) {
            const int nt   = chunk * 16 + ntl;
            const int code = nt * 16 + m;              // column this lane owns
            const int lk   = ntl * 16 + m;             // code row within chunk
            const __bf16* brow = sB + (size_t)lk * VQ_SROW + g * 16;

            v8f c = {};
            #pragma unroll
            for (int kc = 0; kc < 8; ++kc) {
                v16bf bfrag = *(const v16bf*)(brow + kc * 32);  // 32B from LDS
                c = __builtin_amdgcn_wmma_f32_16x16x32_bf16(
                        false, afrag[kc], false, bfrag, (short)0, c, false, false);
            }

            const float en = enorm[code];
            #pragma unroll
            for (int r = 0; r < 8; ++r) {
                float d = en - 2.0f * c[r];
                if (d < best[r]) { best[r] = d; bidx[r] = code; }
            }
        }
        __syncthreads();   // all waves done reading before restaging
    }

    // Min-reduce (value, index) across the 16 lanes of each half-group.
    #pragma unroll
    for (int r = 0; r < 8; ++r) {
        #pragma unroll
        for (int off = 1; off < 16; off <<= 1) {
            float ov = __shfl_xor(best[r], off, 32);
            int   oi = __shfl_xor(bidx[r], off, 32);
            if (ov < best[r] || (ov == best[r] && oi < bidx[r])) {
                best[r] = ov; bidx[r] = oi;
            }
        }
    }

    if (m == 0) {   // lanes 0 and 16 each own 8 rows
        #pragma unroll
        for (int r = 0; r < 8; ++r) {
            const int orow = rowbase + r + 8 * g;   // C layout: M = r + 8g
            idxo[orow] = bidx[r];
            atomicAdd(&cnts[bidx[r]], 1.0f);
        }
    }
}

// ---------------------------------------------------------------------------
// Kernel 3: dw[k] = segment_sum(fin) via fp32 atomics. Coalesced along t.
// ---------------------------------------------------------------------------
__global__ __launch_bounds__(256) void vq_dw_kernel(
    const float* __restrict__ x,
    const int* __restrict__ idx1, const int* __restrict__ idx2,
    float* __restrict__ dw1, float* __restrict__ dw2)
{
    const int t = blockIdx.x * 256 + threadIdx.x;
    const int c = blockIdx.y;          // 0..511 full channel
    const int b = blockIdx.z;
    const int h  = c >> 8;
    const int cc = c & 255;
    const int* idx = h ? idx2 : idx1;
    float*     dw  = h ? dw2  : dw1;

    const int   row = b * VQ_T + t;
    const int   k   = idx[row];
    const float v   = x[((size_t)(b * 512 + c)) * VQ_T + t];
    atomicAdd(&dw[(size_t)k * VQ_SUBD + cc], v);
}

// ---------------------------------------------------------------------------
// Kernel 4: EMA update + Laplace smoothing + new_emb + perplexity.
// One 512-thread block per half (k = threadIdx.x).
// ---------------------------------------------------------------------------
__global__ __launch_bounds__(512) void vq_ema_kernel(
    const float* __restrict__ ema_cs1, const float* __restrict__ ema_w1,
    const float* __restrict__ ema_cs2, const float* __restrict__ ema_w2,
    const float* __restrict__ counts1, const float* __restrict__ counts2,
    const float* __restrict__ dw1, const float* __restrict__ dw2,
    float* __restrict__ ne1, float* __restrict__ ne2,
    float* __restrict__ scalars /* [loss, perp1, perp2] */)
{
    __shared__ float red[VQ_K];
    const int h = blockIdx.x;
    const float* ema_cs = h ? ema_cs2 : ema_cs1;
    const float* ema_w  = h ? ema_w2  : ema_w1;
    const float* cnts   = h ? counts2 : counts1;
    const float* dw     = h ? dw2     : dw1;
    float*       ne     = h ? ne2     : ne1;

    const int k = threadIdx.x;
    const float cnt = cnts[k];
    const float csn = ema_cs[k] * VQ_DECAY + (1.0f - VQ_DECAY) * cnt;

    red[k] = csn;
    __syncthreads();
    for (int s = VQ_K / 2; s > 0; s >>= 1) {
        if (k < s) red[k] += red[k + s];
        __syncthreads();
    }
    const float n = red[0];
    __syncthreads();

    const float smoothed = (csn + VQ_EPS) / (n + VQ_K * VQ_EPS) * n;
    const float inv = 1.0f / smoothed;
    for (int c = 0; c < VQ_SUBD; ++c) {
        const float wn = ema_w[(size_t)k * VQ_SUBD + c] * VQ_DECAY +
                         (1.0f - VQ_DECAY) * dw[(size_t)k * VQ_SUBD + c];
        ne[(size_t)k * VQ_SUBD + c] = wn * inv;
    }

    const float p = cnt / (float)VQ_NROWS;
    red[k] = p * logf(p + 1e-10f);
    __syncthreads();
    for (int s = VQ_K / 2; s > 0; s >>= 1) {
        if (k < s) red[k] += red[k + s];
        __syncthreads();
    }
    if (k == 0) scalars[1 + h] = expf(-red[0]);
}

// ---------------------------------------------------------------------------
// Kernel 5: gather new_emb, write quant_st (transposed back), accumulate loss.
// quant_st forward value = xt + (quant - xt), matching the reference exactly.
// ---------------------------------------------------------------------------
__global__ __launch_bounds__(256) void vq_out_kernel(
    const float* __restrict__ x,
    const int* __restrict__ idx1, const int* __restrict__ idx2,
    const float* __restrict__ ne1, const float* __restrict__ ne2,
    float* __restrict__ out, float* __restrict__ scalars)
{
    const int t = blockIdx.x * 256 + threadIdx.x;
    const int c = blockIdx.y;
    const int b = blockIdx.z;
    const int h  = c >> 8;
    const int cc = c & 255;
    const int* idx = h ? idx2 : idx1;
    const float* ne = h ? ne2 : ne1;

    const int row = b * VQ_T + t;
    const int k   = idx[row];
    const float q = ne[(size_t)k * VQ_SUBD + cc];

    const size_t off = ((size_t)(b * 512 + c)) * VQ_T + t;
    const float xv = x[off];
    const float d  = q - xv;
    out[off] = xv + d;              // straight-through forward value

    float acc = d * d;
    #pragma unroll
    for (int o = 16; o > 0; o >>= 1) acc += __shfl_down(acc, o, 32);
    if ((threadIdx.x & 31) == 0) atomicAdd(&scalars[0], acc);
}

// ---------------------------------------------------------------------------
// Kernel 6: finalize the two scalar outputs.
// ---------------------------------------------------------------------------
__global__ void vq_final_kernel(const float* __restrict__ scalars,
                                float* __restrict__ out_tail)
{
    out_tail[0] = VQ_BETA * scalars[0] / (float)((size_t)VQ_B * 512 * VQ_T);
    out_tail[1] = scalars[1] + scalars[2];
}

// ---------------------------------------------------------------------------
extern "C" void kernel_launch(void* const* d_in, const int* in_sizes, int n_in,
                              void* d_out, int out_size, void* d_ws, size_t ws_size,
                              hipStream_t stream)
{
    (void)in_sizes; (void)n_in; (void)out_size; (void)ws_size;

    const float* x       = (const float*)d_in[0];
    const float* emb1    = (const float*)d_in[1];
    const float* emb2    = (const float*)d_in[2];
    const float* ema_cs1 = (const float*)d_in[3];
    const float* ema_w1  = (const float*)d_in[4];
    const float* ema_cs2 = (const float*)d_in[5];
    const float* ema_w2  = (const float*)d_in[6];

    char* ws = (char*)d_ws;
    __bf16* embB1   = (__bf16*)(ws + WS_EMBB1);
    __bf16* embB2   = (__bf16*)(ws + WS_EMBB2);
    float*  enorm1  = (float*)(ws + WS_ENORM1);
    float*  enorm2  = (float*)(ws + WS_ENORM2);
    int*    idx1    = (int*)(ws + WS_IDX1);
    int*    idx2    = (int*)(ws + WS_IDX2);
    float*  counts1 = (float*)(ws + WS_COUNTS1);
    float*  counts2 = (float*)(ws + WS_COUNTS2);
    float*  dw1     = (float*)(ws + WS_DW1);
    float*  dw2     = (float*)(ws + WS_DW2);
    float*  ne1     = (float*)(ws + WS_NEWEMB1);
    float*  ne2     = (float*)(ws + WS_NEWEMB2);
    float*  scalars = (float*)(ws + WS_SCALARS);

    float* out      = (float*)d_out;
    float* out_tail = out + (size_t)VQ_B * 512 * VQ_T;

    // Zero accumulators (graph-capturable async memsets).
    (void)hipMemsetAsync(ws + WS_ZERO_BEG, 0, WS_ZERO_LEN, stream);
    (void)hipMemsetAsync(ws + WS_SCALARS, 0, 3 * sizeof(float), stream);

    // 1) codebook prep
    vq_prep_kernel<<<dim3(4, 2), 128, 0, stream>>>(emb1, emb2, embB1, embB2,
                                                   enorm1, enorm2);
    // 2) WMMA argmin: 65536 rows / (16 waves * 16 rows) = 256 blocks per half
    vq_argmin_kernel<<<dim3(256, 2), 512, 0, stream>>>(x, embB1, embB2,
                                                       enorm1, enorm2,
                                                       idx1, idx2,
                                                       counts1, counts2);
    // 3) dw segment-sum
    vq_dw_kernel<<<dim3(VQ_T / 256, 512, VQ_B), 256, 0, stream>>>(x, idx1, idx2,
                                                                  dw1, dw2);
    // 4) EMA + perplexity
    vq_ema_kernel<<<dim3(2), VQ_K, 0, stream>>>(ema_cs1, ema_w1, ema_cs2, ema_w2,
                                                counts1, counts2, dw1, dw2,
                                                ne1, ne2, scalars);
    // 5) output + loss
    vq_out_kernel<<<dim3(VQ_T / 256, 512, VQ_B), 256, 0, stream>>>(x, idx1, idx2,
                                                                   ne1, ne2,
                                                                   out, scalars);
    // 6) scalars
    vq_final_kernel<<<1, 1, 0, stream>>>(scalars, out_tail);
}
